// MultiHeadSelfAttention_76957224010355
// MI455X (gfx1250) — matrix operand order repaired and checked by reading, not compiled, not run on hardware
//
#include <hip/hip_runtime.h>
#include <hip/hip_bf16.h>

// ---------------------------------------------------------------------------
// MHA forward for MI455X (gfx1250, wave32, WMMA + TDM + async-LDS).
// Compute-bound (~400 GFLOP vs ~300MB traffic) -> bf16 WMMA with f32 accum.
// GEMM-B tiles: TDM tensor_load_to_lds (TENSORcnt), double buffered.
// Attention K/V tiles: global_load_async_to_lds_b128 (ASYNCcnt), dbl buffered.
// Fragment loads are grouped into register batches so one dep-wait covers a
// batch and WMMAs issue back-to-back.
// ---------------------------------------------------------------------------

typedef __bf16 bf16_t;
typedef __attribute__((ext_vector_type(16))) __bf16 v16bf;
typedef __attribute__((ext_vector_type(8)))  __bf16 v8bf;
typedef __attribute__((ext_vector_type(8)))  float  v8f;
typedef __attribute__((ext_vector_type(4)))  unsigned u32x4;
typedef __attribute__((ext_vector_type(8)))  unsigned u32x8;

#define WMMA_BF16(a, b, c) \
  __builtin_amdgcn_wmma_f32_16x16x32_bf16(false, (a), false, (b), (short)0, (c), false, false)

static constexpr int Bsz    = 4;
static constexpr int Ssz    = 2048;
static constexpr int Dm     = 2048;
static constexpr int Hn     = 16;
static constexpr int Dh     = 128;
static constexpr int ThreeD = 3 * Dm;       // 6144
static constexpr int Mrows  = Bsz * Ssz;    // 8192

// Low 32 bits of a generic pointer to __shared__ == wave-relative LDS offset.
__device__ __forceinline__ unsigned lds_off(const void* p) {
  return (unsigned)(uintptr_t)p;
}

// One async per-lane 16B copy: moves 32 lanes x 16B = 512B per instruction.
__device__ __forceinline__ void async_ld_b128(void* lds_ptr, const void* gptr) {
  asm volatile("global_load_async_to_lds_b128 %0, %1, off"
               :: "v"(lds_off(lds_ptr)),
                  "v"((unsigned long long)(uintptr_t)gptr)
               : "memory");
}

// TDM 2D tile load global->LDS (D# per CDNA5 ISA).
__device__ __forceinline__ void tdm_load_2d(unsigned ldsoff, const void* gaddr,
                                            unsigned tile_d0, unsigned tile_d1,
                                            unsigned tensor_d0,
                                            unsigned tensor_d1,
                                            unsigned stride0_elems) {
  unsigned long long ga = (unsigned long long)(uintptr_t)gaddr;
  u32x4 g0;
  g0[0] = 1u;                                    // count=1 (valid), user mode
  g0[1] = ldsoff;                                // lds_addr
  g0[2] = (unsigned)ga;                          // global_addr[31:0]
  g0[3] = (unsigned)((ga >> 32) & 0x01FFFFFFull) // global_addr[56:32]
          | 0x80000000u;                         // type=2 ("image")
  u32x8 g1;
  g1[0] = 1u << 16;                              // data_size=1 -> 2 bytes
  g1[1] = (tensor_d0 & 0xFFFFu) << 16;           // tensor_dim0[15:0]
  g1[2] = ((tensor_d0 >> 16) & 0xFFFFu) | ((tensor_d1 & 0xFFFFu) << 16);
  g1[3] = ((tensor_d1 >> 16) & 0xFFFFu) | (tile_d0 << 16);  // tile_dim0
  g1[4] = tile_d1;                               // tile_dim1 (tile_dim2=0)
  g1[5] = stride0_elems;                         // tensor_dim0_stride[31:0]
  g1[6] = 0u;
  g1[7] = 0u;
  asm volatile("tensor_load_to_lds %0, %1" :: "s"(g0), "s"(g1) : "memory");
}

// ---------------------------------------------------------------------------
// f32 -> bf16 elementwise convert
// ---------------------------------------------------------------------------
__global__ void cvt_f32_bf16(const float* __restrict__ src,
                             bf16_t* __restrict__ dst, int n) {
  int i = blockIdx.x * blockDim.x + threadIdx.x;
  if (i < n) dst[i] = (bf16_t)src[i];
}

// ---------------------------------------------------------------------------
// NT GEMM: C[M,N] = A[M,K] * B[N,K]^T.  Block = 256 threads = 8 waves,
// block tile 256x64, wave tile 32x64 (2 M-sub x 4 N-sub = 8 WMMA / K-step),
// K-step 32.  The 64x32 B tile is shared by all 8 waves -> TDM into LDS,
// double buffered.  A/B fragments use contiguous k = 16*(lane/16) + s.
// ---------------------------------------------------------------------------
template <bool OUT_BF16>
__global__ __launch_bounds__(256)
void gemm_nt_wmma(const bf16_t* __restrict__ A, const bf16_t* __restrict__ Bm,
                  void* __restrict__ Cv, int M, int N, int K) {
  __shared__ __align__(16) bf16_t btile[2][64 * 32];

  const int lane  = threadIdx.x & 31;
  const int wave  = threadIdx.x >> 5;
  const int l15   = lane & 15;
  const int khalf = lane >> 4;
  const int mbase = blockIdx.y * 256 + wave * 32;
  const int nbase = blockIdx.x * 64;

  const bf16_t* bsrc  = Bm + (size_t)nbase * K;
  const int     niter = K / 32;

  if (wave == 0)
    tdm_load_2d(lds_off(&btile[0][0]), bsrc, 32, 64,
                (unsigned)K, (unsigned)N, (unsigned)K);

  v8f acc[2][4] = {};
  const bf16_t* arow[2] = {A + (size_t)(mbase + l15) * K,
                           A + (size_t)(mbase + 16 + l15) * K};

  for (int i = 0; i < niter; ++i) {
    const int k0  = i * 32;
    const int buf = i & 1;
    if (wave == 0 && i + 1 < niter)
      tdm_load_2d(lds_off(&btile[buf ^ 1][0]), bsrc + k0 + 32, 32, 64,
                  (unsigned)K, (unsigned)N, (unsigned)K);

    // two A fragments from global (overlap the TDM wait)
    v16bf af[2];
#pragma unroll
    for (int ms = 0; ms < 2; ++ms) {
      v8bf a0 = *(const v8bf*)(arow[ms] + k0 + khalf * 16);
      v8bf a1 = *(const v8bf*)(arow[ms] + k0 + khalf * 16 + 8);
#pragma unroll
      for (int e = 0; e < 8; ++e) { af[ms][e] = a0[e]; af[ms][8 + e] = a1[e]; }
    }

    if (wave == 0) {
      if (i + 1 < niter) __builtin_amdgcn_s_wait_tensorcnt(1);
      else               __builtin_amdgcn_s_wait_tensorcnt(0);
    }
    __syncthreads();  // tile i visible to all waves

    // batch all 4 B fragments, then 8 back-to-back WMMAs
    v16bf bfq[4];
#pragma unroll
    for (int t = 0; t < 4; ++t)
      bfq[t] = *(const v16bf*)&btile[buf][(t * 16 + l15) * 32 + khalf * 16];
#pragma unroll
    for (int t = 0; t < 4; ++t) acc[0][t] = WMMA_BF16(af[0], bfq[t], acc[0][t]);
#pragma unroll
    for (int t = 0; t < 4; ++t) acc[1][t] = WMMA_BF16(af[1], bfq[t], acc[1][t]);

    __syncthreads();  // done reading buf before it is overwritten
  }

#pragma unroll
  for (int ms = 0; ms < 2; ++ms)
#pragma unroll
    for (int t = 0; t < 4; ++t)
#pragma unroll
      for (int j = 0; j < 8; ++j) {
        const int row = mbase + ms * 16 + j + 8 * khalf;
        const int col = nbase + t * 16 + l15;
        if (OUT_BF16)
          ((bf16_t*)Cv)[(size_t)row * N + col] = (bf16_t)acc[ms][t][j];
        else
          ((float*)Cv)[(size_t)row * N + col] = acc[ms][t][j];
      }
}

// ---------------------------------------------------------------------------
// RoPE + head split. qkv bf16 [B,S,6144] -> q,k [B,H,S,Dh] (q pre-scaled by
// 1/sqrt(Dh)), v transposed to [B,H,Dh,S] for contiguous PV B-fragments.
// ---------------------------------------------------------------------------
__global__ void rope_split(const bf16_t* __restrict__ qkv,
                           bf16_t* __restrict__ q, bf16_t* __restrict__ k,
                           bf16_t* __restrict__ vT) {
  const int idx = blockIdx.x * blockDim.x + threadIdx.x;  // (b,h,s,i)
  const int i = idx & 63;
  const int s = (idx >> 6) & (Ssz - 1);
  const int h = (idx >> 17) & (Hn - 1);
  const int b = idx >> 21;
  if (b >= Bsz) return;

  const size_t base = ((size_t)(b * Ssz + s)) * ThreeD + h * Dh + 2 * i;
  const float q0 = (float)qkv[base + 0];
  const float q1 = (float)qkv[base + 1];
  const float k0 = (float)qkv[base + Dm];
  const float k1 = (float)qkv[base + Dm + 1];
  const float v0 = (float)qkv[base + 2 * Dm];
  const float v1 = (float)qkv[base + 2 * Dm + 1];

  const float inv = __expf(-(float)(2 * i) * (9.210340371976184f / 128.0f));
  float sn, cs;
  __sincosf((float)s * inv, &sn, &cs);
  const float qscale = 0.08838834764831845f;  // 1/sqrt(128)

  const size_t qkidx = ((size_t)((b * Hn + h) * Ssz + s)) * Dh + 2 * i;
  q[qkidx]     = (bf16_t)((q0 * cs - q1 * sn) * qscale);
  q[qkidx + 1] = (bf16_t)((q0 * sn + q1 * cs) * qscale);
  k[qkidx]     = (bf16_t)(k0 * cs - k1 * sn);
  k[qkidx + 1] = (bf16_t)(k0 * sn + k1 * cs);

  const size_t vtb = ((size_t)(b * Hn + h)) * Dh * Ssz;
  vT[vtb + (size_t)(2 * i) * Ssz + s]     = (bf16_t)v0;
  vT[vtb + (size_t)(2 * i + 1) * Ssz + s] = (bf16_t)v1;
}

// ---------------------------------------------------------------------------
// Flash attention. Block = 128 threads (4 waves) on one (b, h, 64-query tile);
// each wave owns 16 query rows.  K (32x128) and V^T (128x32) tiles shared by
// all 4 waves -> double-buffered async-LDS copies (ASYNCcnt).  P re-laid out
// C-layout -> A-layout through a per-wave LDS bounce (DScnt).  Causal mask
// VALU only on the (wave-uniform) boundary tiles.
// ---------------------------------------------------------------------------
__global__ __launch_bounds__(128)
void flash_attn(const bf16_t* __restrict__ Q, const bf16_t* __restrict__ Km,
                const bf16_t* __restrict__ VT, bf16_t* __restrict__ Aout) {
  __shared__ __align__(16) bf16_t ktile[2][32 * 128];  // [key][d]
  __shared__ __align__(16) bf16_t vtile[2][128 * 32];  // [d][key]
  __shared__ __align__(64) bf16_t ptile[4][16 * 32];   // per-wave P bounce

  const int tid   = threadIdx.x;
  const int lane  = tid & 31;
  const int wave  = tid >> 5;
  const int l15   = lane & 15;
  const int khalf = lane >> 4;

  const int b  = blockIdx.z;
  const int h  = blockIdx.y;
  const int q0 = blockIdx.x * 64;
  const int qw = q0 + wave * 16;

  const size_t bh = (size_t)(b * Hn + h);
  const bf16_t* Qp = Q + bh * Ssz * Dh;
  const bf16_t* Kp = Km + bh * Ssz * Dh;
  const bf16_t* Vp = VT + bh * (size_t)Dh * Ssz;

  // 8 async b128 per wave per tile: 4 for K (8KB/block) + 4 for V^T (8KB).
  auto copy_tiles = [&](int buf, int kb) {
#pragma unroll
    for (int i = 0; i < 4; ++i) {
      const int c = tid + i * 128;           // 16B chunk id, 512 per tile
      const int key = c >> 4, dc = c & 15;   // K: [key][d] rows of 128
      async_ld_b128(&ktile[buf][c * 8], Kp + (size_t)(kb + key) * Dh + dc * 8);
    }
#pragma unroll
    for (int i = 0; i < 4; ++i) {
      const int c = tid + i * 128;
      const int d = c >> 2, kc = c & 3;      // V^T: [d][key] rows of 32
      async_ld_b128(&vtile[buf][c * 8], Vp + (size_t)d * Ssz + kb + kc * 8);
    }
  };

  copy_tiles(0, 0);

  // Q fragments: 4 chunks of 16x32 over head dim (A-layout), from global.
  v16bf qf[4];
  {
    const bf16_t* qrow = Qp + (size_t)(qw + l15) * Dh;
#pragma unroll
    for (int c = 0; c < 4; ++c) {
      v8bf a0 = *(const v8bf*)(qrow + c * 32 + khalf * 16);
      v8bf a1 = *(const v8bf*)(qrow + c * 32 + khalf * 16 + 8);
#pragma unroll
      for (int e = 0; e < 8; ++e) { qf[c][e] = a0[e]; qf[c][8 + e] = a1[e]; }
    }
  }

  v8f oacc[8] = {};
  float mrun[8], lrun[8];
#pragma unroll
  for (int j = 0; j < 8; ++j) { mrun[j] = -INFINITY; lrun[j] = 0.f; }

  const int ntiles = (q0 + 64) / 32;  // causal
  for (int t = 0; t < ntiles; ++t) {
    const int kb  = t * 32;
    const int buf = t & 1;
    if (t + 1 < ntiles) {
      copy_tiles(buf ^ 1, kb + 32);
      asm volatile("s_wait_asynccnt 8" ::: "memory");  // tile t done
    } else {
      asm volatile("s_wait_asynccnt 0" ::: "memory");
    }
    __syncthreads();

    // ---- S = Q K^T : two 16x16 subtiles; batch 4 K frags then 4 WMMAs ----
    v8f sacc[2] = {};
#pragma unroll
    for (int ns = 0; ns < 2; ++ns) {
      const bf16_t* krow = &ktile[buf][(ns * 16 + l15) * 128 + khalf * 16];
      v16bf kf[4];
#pragma unroll
      for (int c = 0; c < 4; ++c) kf[c] = *(const v16bf*)(krow + c * 32);
#pragma unroll
      for (int c = 0; c < 4; ++c) sacc[ns] = WMMA_BF16(qf[c], kf[c], sacc[ns]);
    }

    // ---- causal mask (boundary tiles only; wave-uniform) + softmax ----
    const bool need_mask = (kb + 31) > qw;
    float alpha[8];
#pragma unroll
    for (int j = 0; j < 8; ++j) {
      const int qrow = qw + j + 8 * khalf;
      float s0 = sacc[0][j];
      float s1 = sacc[1][j];
      if (need_mask) {
        if (kb + l15 > qrow)      s0 = -INFINITY;
        if (kb + 16 + l15 > qrow) s1 = -INFINITY;
      }

      float rm = fmaxf(s0, s1);
#pragma unroll
      for (int off = 8; off >= 1; off >>= 1)
        rm = fmaxf(rm, __shfl_xor(rm, off, 32));

      const float mnew = fmaxf(mrun[j], rm);
      const float a    = __expf(mrun[j] - mnew);
      const float p0   = __expf(s0 - mnew);
      const float p1   = __expf(s1 - mnew);
      float rs = p0 + p1;
#pragma unroll
      for (int off = 8; off >= 1; off >>= 1)
        rs += __shfl_xor(rs, off, 32);

      mrun[j]  = mnew;
      lrun[j]  = lrun[j] * a + rs;
      alpha[j] = a;

      ptile[wave][(j + 8 * khalf) * 32 + l15]      = (bf16_t)p0;
      ptile[wave][(j + 8 * khalf) * 32 + 16 + l15] = (bf16_t)p1;
    }

#pragma unroll
    for (int d = 0; d < 8; ++d)
#pragma unroll
      for (int j = 0; j < 8; ++j) oacc[d][j] *= alpha[j];

    // wave-local LDS RAW (DS ops in-order per wave)
    asm volatile("s_wait_dscnt 0" ::: "memory");

    v16bf pf;  // P in A-layout
    {
      const bf16_t* prow = &ptile[wave][l15 * 32];
      v8bf a0 = *(const v8bf*)(prow + khalf * 16);
      v8bf a1 = *(const v8bf*)(prow + khalf * 16 + 8);
#pragma unroll
      for (int e = 0; e < 8; ++e) { pf[e] = a0[e]; pf[8 + e] = a1[e]; }
    }

    // ---- O += P V : 8 subtiles; batch V frags in groups of 4 ----
#pragma unroll
    for (int dd = 0; dd < 8; dd += 4) {
      v16bf vfq[4];
#pragma unroll
      for (int e = 0; e < 4; ++e)
        vfq[e] =
            *(const v16bf*)&vtile[buf][((dd + e) * 16 + l15) * 32 + khalf * 16];
#pragma unroll
      for (int e = 0; e < 4; ++e)
        oacc[dd + e] = WMMA_BF16(pf, vfq[e], oacc[dd + e]);
    }
    __syncthreads();  // done reading buf before next copy overwrites it
  }

  // ---- epilogue: O /= l ; write bf16 A [B,S,H*Dh] ----
  float invl[8];
#pragma unroll
  for (int j = 0; j < 8; ++j) invl[j] = 1.0f / lrun[j];
#pragma unroll
  for (int d = 0; d < 8; ++d)
#pragma unroll
    for (int j = 0; j < 8; ++j) {
      const int row = qw + j + 8 * khalf;
      const int col = h * Dh + d * 16 + l15;
      Aout[(size_t)(b * Ssz + row) * Dm + col] = (bf16_t)(oacc[d][j] * invl[j]);
    }
}

// ---------------------------------------------------------------------------
// Launch
// ---------------------------------------------------------------------------
extern "C" void kernel_launch(void* const* d_in, const int* in_sizes, int n_in,
                              void* d_out, int out_size, void* d_ws,
                              size_t ws_size, hipStream_t stream) {
  const float* x     = (const float*)d_in[0];
  const float* w_qkv = (const float*)d_in[1];
  const float* w_out = (const float*)d_in[2];
  float* out = (float*)d_out;

  char* ws = (char*)d_ws;
  size_t off = 0;
  auto alloc = [&](size_t bytes) -> void* {
    void* p = ws + off;
    off += (bytes + 255) & ~(size_t)255;
    return p;
  };

  bf16_t* x16    = (bf16_t*)alloc((size_t)Mrows * Dm * 2);
  bf16_t* wqkv16 = (bf16_t*)alloc((size_t)ThreeD * Dm * 2);
  bf16_t* wout16 = (bf16_t*)alloc((size_t)Dm * Dm * 2);
  bf16_t* qkv16  = (bf16_t*)alloc((size_t)Mrows * ThreeD * 2);
  bf16_t* q16    = (bf16_t*)alloc((size_t)Bsz * Hn * Ssz * Dh * 2);
  bf16_t* k16    = (bf16_t*)alloc((size_t)Bsz * Hn * Ssz * Dh * 2);
  bf16_t* vt16   = (bf16_t*)alloc((size_t)Bsz * Hn * Dh * Ssz * 2);
  bf16_t* a16    = (bf16_t*)alloc((size_t)Mrows * Dm * 2);

  int nx = Mrows * Dm, nq = ThreeD * Dm, no = Dm * Dm;
  cvt_f32_bf16<<<(nx + 255) / 256, 256, 0, stream>>>(x, x16, nx);
  cvt_f32_bf16<<<(nq + 255) / 256, 256, 0, stream>>>(w_qkv, wqkv16, nq);
  cvt_f32_bf16<<<(no + 255) / 256, 256, 0, stream>>>(w_out, wout16, no);

  gemm_nt_wmma<true><<<dim3(ThreeD / 64, Mrows / 256), 256, 0, stream>>>(
      x16, wqkv16, qkv16, Mrows, ThreeD, Dm);

  rope_split<<<(Bsz * Hn * Ssz * 64) / 256, 256, 0, stream>>>(qkv16, q16, k16,
                                                              vt16);

  flash_attn<<<dim3(Ssz / 64, Hn, Bsz), 128, 0, stream>>>(q16, k16, vt16, a16);

  gemm_nt_wmma<false><<<dim3(Dm / 64, Mrows / 256), 256, 0, stream>>>(
      a16, wout16, out, Mrows, Dm, Dm);
}
